// VQGANCodebook_34531537060173
// MI455X (gfx1250) — compile-verified, matrix operand order checked by reading, not run on hardware
//
#include <hip/hip_runtime.h>
#include <hip/hip_bf16.h>
#include <math.h>

typedef __attribute__((ext_vector_type(16))) _Float16 v16h;
typedef __attribute__((ext_vector_type(8)))  _Float16 v8h;
typedef __attribute__((ext_vector_type(8)))  float    v8f;
typedef __attribute__((ext_vector_type(4)))  unsigned v4u;
typedef __attribute__((ext_vector_type(8)))  int      v8i;
typedef __attribute__((ext_vector_type(4)))  int      v4i;

#define N_TOT   131072          // B*H*W
#define C_DIM   256
#define K_CODES 1024
#define HW      4096            // H*W
#define TILE_M  128
#define LDS_STRIDE 264          // 256 + 8 halves padding (bank-conflict break)
#define A_BYTES (TILE_M * LDS_STRIDE * 2)       // 67584
#define B_ROW_HALVES 264                        // 512B data + 16B TDM pad
#define B_BUF_BYTES (16 * B_ROW_HALVES * 2)     // 8448
#define SMEM_BYTES (A_BYTES + 2 * B_BUF_BYTES)  // 84480

// ---------------------------------------------------------------------------
// TDM: DMA a 16-code x 256-ch f16 tile of the codebook into LDS, padding each
// 512B row with 16B so ds_load_b128 B-fragment reads are bank-conflict free.
// D# layout per CDNA5 ISA §8 (group0: count/lds_addr/global_addr/type,
// group1: data_size + pad ctrl + tensor/tile dims + stride). 2D tensor ->
// groups 2/3 zero. Tracked by TENSORcnt.
// ---------------------------------------------------------------------------
__device__ __forceinline__ void tdm_load_b_tile(const _Float16* gsrc,
                                                unsigned ldsAddr)
{
    unsigned long long ga = (unsigned long long)(size_t)gsrc;
    v4u g0;
    g0[0] = 1u;                                   // count=1, user descriptor
    g0[1] = ldsAddr;                              // lds_addr (bytes)
    g0[2] = (unsigned)(ga & 0xFFFFFFFFu);         // global_addr[31:0]
    g0[3] = (unsigned)((ga >> 32) & 0x01FFFFFFu)  // global_addr[56:32]
          | (2u << 30);                           // type = 2 ("image")
    v8i g1;
    g1[0] = (int)((1u << 16)    // data_size = 1 -> 2 bytes/elem
                | (1u << 20)    // pad_enable
                | (6u << 22)    // pad_interval: 128 dwords (one 512B row)
                | (3u << 25));  // pad_amount: 4 dwords (16B) per row
    g1[1] = (int)(256u << 16);  // tensor_dim0 = 256  (bits 79:48, low half)
    g1[2] = (int)(1024u << 16); // dim0 hi = 0 ; tensor_dim1 = 1024 (lo16)
    g1[3] = (int)(256u << 16);  // dim1 hi = 0 ; tile_dim0 = 256
    g1[4] = 16;                 // tile_dim1 = 16 ; tile_dim2 = 0
    g1[5] = 256;                // tensor_dim0_stride = 256 elems (lo32)
    g1[6] = 0;                  // stride0 hi ; tensor_dim1_stride lo
    g1[7] = 0;
    v4i gz = {0, 0, 0, 0};
#if defined(__clang_major__) && (__clang_major__ >= 23)
    v8i gz8 = {0, 0, 0, 0, 0, 0, 0, 0};
    __builtin_amdgcn_tensor_load_to_lds(g0, g1, gz, gz, gz8, 0);
#else
    __builtin_amdgcn_tensor_load_to_lds(g0, g1, gz, gz, 0);
#endif
}

// ---------------------------------------------------------------------------
// Kernel 0: codebook prep — fp32 emb [K,C] -> f16 copy + per-code |e|^2
// ---------------------------------------------------------------------------
__global__ __launch_bounds__(256) void vq_prep_emb(
    const float* __restrict__ emb, _Float16* __restrict__ embh,
    float* __restrict__ embnorm)
{
    const int k = blockIdx.x;
    const int c = threadIdx.x;
    float v = emb[k * C_DIM + c];
    embh[k * C_DIM + c] = (_Float16)v;
    __shared__ float red[256];
    red[c] = v * v;
    __syncthreads();
    for (int s = 128; s > 0; s >>= 1) {
        if (c < s) red[c] += red[c + s];
        __syncthreads();
    }
    if (c == 0) embnorm[k] = red[0];
}

// ---------------------------------------------------------------------------
// Kernel 1: zero the accumulators (ws is not re-poisoned between replays)
// ---------------------------------------------------------------------------
__global__ void vq_init(unsigned* __restrict__ hist, float* __restrict__ sumsq)
{
    hist[threadIdx.x] = 0u;
    if (threadIdx.x == 0) *sumsq = 0.0f;
}

// ---------------------------------------------------------------------------
// Kernel 2: WMMA distance GEMM + fused per-row argmin over all K codes.
// 256 threads = 8 waves; tile = 128 rows; each wave owns 16 rows with its
// A fragments resident in registers. B chunks (16 codes x 256 ch) are DMA'd
// into a double-buffered LDS tile by the Tensor Data Mover (wave 0), with
// the next DMA overlapping the current 8-WMMA chain.
// dist = |e|^2 - 2*(z.e)   (|z|^2 constant per row -> irrelevant for argmin)
// ---------------------------------------------------------------------------
__global__ __launch_bounds__(256) void vq_argmin(
    const float*    __restrict__ z_e,      // [B,C,H,W] fp32
    const _Float16* __restrict__ embh,     // [K,C] f16
    const float*    __restrict__ embnorm,  // [K]
    int*            __restrict__ zidx,     // [N]
    float*          __restrict__ outz)     // [N] (indices as float, to d_out)
{
    extern __shared__ char smem[];
    _Float16* ldsA = (_Float16*)smem;                // 128 x 264 halves
    _Float16* ldsB = (_Float16*)(smem + A_BYTES);    // 2 x (16 x 264 halves)
    const unsigned ldsBAddr0 = (unsigned)(size_t)(smem + A_BYTES);

    const int t       = threadIdx.x;
    const int rowBase = blockIdx.x * TILE_M;         // multiple of 128
    const int bImg    = rowBase >> 12;               // HW = 4096
    const int hw0     = rowBase & (HW - 1);          // tile inside one image
    const size_t zbase = (size_t)bImg * C_DIM * HW + (size_t)hw0;

    // Kick the first B-chunk DMA before staging A so it overlaps.
    if (t < 32) tdm_load_b_tile(embh, ldsBAddr0);

    // ---- Stage 1: stream 128x256 fp32 tile -> f16 LDS (coalesced per plane)
    #pragma unroll 4
    for (int iter = 0; iter < 128; ++iter) {
        int flat = t + iter * 256;
        int c = flat >> 7;           // 0..255
        int r = flat & 127;          // 0..127 contiguous hw
        float v = z_e[zbase + (size_t)c * HW + (size_t)r];
        ldsA[r * LDS_STRIDE + c] = (_Float16)v;
    }
    if (t < 32) __builtin_amdgcn_s_wait_tensorcnt(0);
    __syncthreads();

    const int lane  = t & 31;
    const int wave  = t >> 5;
    const int M0    = wave * 16;                    // rows of this wave
    const int Mlane = M0 + (lane & 15);
    const int K0    = (lane < 16) ? 0 : 8;          // ISA 16-bit A/B layout
    const int colLane = lane & 15;

    // ---- Stage 2: preload the 8 A fragments (K = 256 in chunks of 32)
    v16h afrag[8];
    #pragma unroll
    for (int cc = 0; cc < 8; ++cc) {
        const _Float16* p = &ldsA[Mlane * LDS_STRIDE + cc * 32 + K0];
        v8h lo = *(const v8h*)(p);          // K = base+0..7
        v8h hi = *(const v8h*)(p + 16);     // K = base+16..23
        v16h a;
        #pragma unroll
        for (int i = 0; i < 8; ++i) { a[i] = lo[i]; a[i + 8] = hi[i]; }
        afrag[cc] = a;
    }

    float bestv[8];
    int   besti[8];
    #pragma unroll
    for (int r = 0; r < 8; ++r) { bestv[r] = 3.0e38f; besti[r] = 0; }

    // ---- Stage 3: sweep all 1024 codes, 16 at a time, TDM double-buffered
    int bi = 0;
    for (int kb = 0; kb < K_CODES; kb += 16) {
        // DMA next chunk into the other buffer while we compute this one
        if (t < 32 && (kb + 16) < K_CODES)
            tdm_load_b_tile(embh + (size_t)(kb + 16) * C_DIM,
                            ldsBAddr0 + (unsigned)((bi ^ 1) * B_BUF_BYTES));

        float en = embnorm[kb + colLane];

        // Batch all 16 ds_load_b128 B-fragment reads so they pipeline
        const _Float16* bbase =
            ldsB + bi * (B_BUF_BYTES / 2) + colLane * B_ROW_HALVES + K0;
        v16h bfrag[8];
        #pragma unroll
        for (int cc = 0; cc < 8; ++cc) {
            const _Float16* pb = bbase + cc * 32;
            v8h lo = *(const v8h*)(pb);
            v8h hi = *(const v8h*)(pb + 16);
            v16h b;
            #pragma unroll
            for (int i = 0; i < 8; ++i) { b[i] = lo[i]; b[i + 8] = hi[i]; }
            bfrag[cc] = b;
        }
        v8f acc = {};
        #pragma unroll
        for (int cc = 0; cc < 8; ++cc) {
            acc = __builtin_amdgcn_wmma_f32_16x16x32_f16(
                false, afrag[cc], false, bfrag[cc], (short)0, acc,
                false, false);
        }

        const int col = kb + colLane;
        #pragma unroll
        for (int r = 0; r < 8; ++r) {
            float dist = en - 2.0f * acc[r];
            if (dist < bestv[r]) { bestv[r] = dist; besti[r] = col; }
        }

        if (t < 32) __builtin_amdgcn_s_wait_tensorcnt(0);
        __syncthreads();             // publish next buffer / retire this one
        bi ^= 1;
    }

    // ---- Stage 4: butterfly argmin across the 16 column lanes
    #pragma unroll
    for (int mask = 1; mask <= 8; mask <<= 1) {
        #pragma unroll
        for (int r = 0; r < 8; ++r) {
            float ov = __shfl_xor(bestv[r], mask, 32);
            int   oi = __shfl_xor(besti[r], mask, 32);
            if (ov < bestv[r] || (ov == bestv[r] && oi < besti[r])) {
                bestv[r] = ov; besti[r] = oi;
            }
        }
    }
    if ((lane & 15) == 0) {
        const int rb = M0 + ((lane >= 16) ? 8 : 0);   // C-layout: VGPR r -> M
        #pragma unroll
        for (int r = 0; r < 8; ++r) {
            int row = rowBase + rb + r;
            zidx[row] = besti[r];
            outz[row] = (float)besti[r];
        }
    }
}

// ---------------------------------------------------------------------------
// Kernel 3: gather z_q (== forward value of z_q_st), fused MSE + histogram.
// Block handles 64 consecutive n (contiguous hw within one b) x all 256 ch.
// ---------------------------------------------------------------------------
__global__ __launch_bounds__(256) void vq_gather(
    const float* __restrict__ z_e, const float* __restrict__ emb,
    const int*   __restrict__ zidx, float* __restrict__ out_zq,
    unsigned*    __restrict__ hist, float* __restrict__ sumsq)
{
    const int t     = threadIdx.x;
    const int nBase = blockIdx.x * 64;
    const int bImg  = nBase >> 12;
    const int hw0   = nBase & (HW - 1);
    const int nOff  = t & 63;
    const int c0    = t >> 6;               // 0..3
    const int idx   = zidx[nBase + nOff];
    const size_t base = (size_t)bImg * C_DIM * HW + (size_t)(hw0 + nOff);

    float sum = 0.0f;
    #pragma unroll 8
    for (int c = c0; c < C_DIM; c += 4) {
        float ze = z_e[base + (size_t)c * HW];
        float q  = emb[idx * C_DIM + c];
        float d  = ze - q;
        sum += d * d;
        out_zq[base + (size_t)c * HW] = q;
    }
    __shared__ float red[256];
    red[t] = sum;
    __syncthreads();
    for (int s = 128; s > 0; s >>= 1) {
        if (t < s) red[t] += red[t + s];
        __syncthreads();
    }
    if (t == 0) atomicAdd(sumsq, red[0]);
    if (t < 64) atomicAdd(&hist[idx], 1u);   // threads 0..63 hold distinct n
}

// ---------------------------------------------------------------------------
// Kernel 4: finalize — perplexity from histogram, vq_loss = (1+beta)*MSE
// ---------------------------------------------------------------------------
__global__ __launch_bounds__(1024) void vq_finalize(
    const unsigned* __restrict__ hist, const float* __restrict__ sumsq,
    const float* __restrict__ beta, float* __restrict__ outLoss,
    float* __restrict__ outPerp)
{
    __shared__ float red[1024];
    const int t = threadIdx.x;
    float p = (float)hist[t] * (1.0f / (float)N_TOT);
    red[t] = p * logf(p + 1e-10f);
    __syncthreads();
    for (int s = 512; s > 0; s >>= 1) {
        if (t < s) red[t] += red[t + s];
        __syncthreads();
    }
    if (t == 0) {
        *outPerp = expf(-red[0]);
        float mse = *sumsq * (1.0f / ((float)N_TOT * (float)C_DIM));
        *outLoss = (1.0f + *beta) * mse;   // commitment + e2z are same MSE fwd
    }
}

// ---------------------------------------------------------------------------
extern "C" void kernel_launch(void* const* d_in, const int* in_sizes, int n_in,
                              void* d_out, int out_size, void* d_ws, size_t ws_size,
                              hipStream_t stream) {
    const float* z_e  = (const float*)d_in[0];   // [32,256,64,64]
    const float* emb  = (const float*)d_in[1];   // [1024,256]
    const float* beta = (const float*)d_in[2];   // scalar

    // workspace carve-up
    char* ws = (char*)d_ws;
    _Float16* embh    = (_Float16*)(ws);                          // 512 KB
    float*    embnorm = (float*)(ws + 512 * 1024);                // 4 KB
    int*      zidx    = (int*)(ws + 512 * 1024 + 4096);           // 512 KB
    unsigned* hist    = (unsigned*)(ws + 1024 * 1024 + 4096);     // 4 KB
    float*    sumsq   = (float*)(ws + 1024 * 1024 + 8192);        // 4 B

    // d_out: [vq_loss(1) | z_q_st(32*256*64*64) | perplexity(1) | z(131072)]
    float* out     = (float*)d_out;
    float* outLoss = out;
    float* outZq   = out + 1;
    float* outPerp = out + 1 + (size_t)N_TOT * C_DIM;
    float* outZ    = out + 2 + (size_t)N_TOT * C_DIM;

    vq_prep_emb<<<K_CODES, 256, 0, stream>>>(emb, embh, embnorm);
    vq_init<<<1, K_CODES, 0, stream>>>(hist, sumsq);
    vq_argmin<<<N_TOT / TILE_M, 256, SMEM_BYTES, stream>>>(
        z_e, embh, embnorm, zidx, outZ);
    vq_gather<<<N_TOT / 64, 256, 0, stream>>>(z_e, emb, zidx, outZq, hist, sumsq);
    vq_finalize<<<1, 1024, 0, stream>>>(hist, sumsq, beta, outLoss, outPerp);
}
